// SelfAttentionHead_60902636257854
// MI455X (gfx1250) — matrix operand order repaired and checked by reading, not compiled, not run on hardware
//
#include <hip/hip_runtime.h>

// ---------------------------------------------------------------------------
// Fused self-attention head for MI455X (gfx1250, wave32, WMMA + TDM).
//   B=8, S=2048, D=1024, fp32 in/out, bf16 WMMA internally (f32 accumulate).
//
// Kernel 0: one-time fp32 -> bf16 conversion of src and weights.
// Kernel 1: fused Q/K/V projection GEMM, 32x16 output tile per wave
//           (6 v_wmma per k-step, pure b128 bf16 loads).
//           V stored transposed so attention B-fragments are contiguous.
// Kernel 2: flash-attention per 16-query block; the 32x1024 bf16 K tile for
//           each key block is staged into LDS by the Tensor Data Mover
//           (tensor_load_to_lds, double-buffered one block ahead, TENSORcnt
//           synchronized); 8 waves split the D contraction for scores
//           (LDS ds_add_f32 reduction), online softmax in LDS, P*V WMMAs
//           per wave over a 128-column slice of transposed V.
// ---------------------------------------------------------------------------

typedef __attribute__((ext_vector_type(16))) __bf16 v16bf;
typedef __attribute__((ext_vector_type(8)))  __bf16 bf16x8;
typedef __attribute__((ext_vector_type(8)))  float  v8f;
typedef __attribute__((ext_vector_type(8)))  float  f32x8;
typedef __attribute__((ext_vector_type(4)))  unsigned int u32x4;
typedef __attribute__((ext_vector_type(8)))  int    i32x8;
typedef __attribute__((ext_vector_type(4)))  int    i32x4;

#define D_MODEL 1024
#define SEQ     2048
#define BATCH   8
#define NEG_BIG (-1e9f)
#define ATT_SCALE 0.03125f   /* 1/sqrt(1024) */
#define KTILE_BYTES (32 * D_MODEL * 2)   /* one 32x1024 bf16 K tile = 64 KB */

static __device__ __forceinline__ v8f wmma_bf16(v16bf a, v16bf b, v8f c) {
  // D = A(16x32) * B(32x16) + C, f32 accumulate
  return __builtin_amdgcn_wmma_f32_16x16x32_bf16(
      /*neg_a=*/false, a, /*neg_b=*/false, b,
      /*c_mod=*/(short)0, c, /*reuse_a=*/false, /*reuse_b=*/false);
}

// A-fragment (16x32 bf16, M x K): lane m holds row m; element e -> K index
// k = e + 16*hi  => 16 contiguous halfs.
static __device__ __forceinline__ v16bf a_frag_bf16(const __bf16* row, int d0, int hi) {
  bf16x8 c0 = *(const bf16x8*)(row + d0 + 16 * hi);
  bf16x8 c1 = *(const bf16x8*)(row + d0 + 16 * hi + 8);
  v16bf r;
#pragma unroll
  for (int i = 0; i < 8; ++i) { r[i] = c0[i]; r[i + 8] = c1[i]; }
  return r;
}

// B-fragment (32x16 bf16, K x N): lane n holds col n; element e -> K index
// k = e + 16*hi  => 16 contiguous halfs starting at p (=base + d0 + 16*hi).
static __device__ __forceinline__ v16bf b_frag_bf16(const __bf16* p) {
  bf16x8 c0 = *(const bf16x8*)(p);
  bf16x8 c1 = *(const bf16x8*)(p + 8);
  v16bf r;
#pragma unroll
  for (int i = 0; i < 8; ++i) { r[i] = c0[i]; r[i + 8] = c1[i]; }
  return r;
}

// ---------------------------------------------------------------------------
// Kernel 0: bulk fp32 -> bf16 conversion (8 elements / thread).
// ---------------------------------------------------------------------------
__global__ __launch_bounds__(256) void cvt_bf16_kernel(
    const float* __restrict__ in, __bf16* __restrict__ out, long n)
{
  const long i = ((long)blockIdx.x * 256 + threadIdx.x) * 8;
  if (i + 8 <= n) {
    f32x8 x = *(const f32x8*)(in + i);
    bf16x8 y;
#pragma unroll
    for (int j = 0; j < 8; ++j) y[j] = (__bf16)x[j];
    *(bf16x8*)(out + i) = y;
  }
}

// ---------------------------------------------------------------------------
// Kernel 1: fused QKV projection (all-bf16 operands).
// Grid: (B*S/32) * (D/16/8) blocks of 256 threads; each wave computes a
// 32x16 tile of Q, K and V (A fragments reused 3x, B fragments reused 2x).
// Q,K stored row-major bf16 [b][s][d]; V stored transposed bf16 [b][d][s].
// ---------------------------------------------------------------------------
__global__ __launch_bounds__(256) void qkv_proj_kernel(
    const __bf16* __restrict__ Xb,
    const __bf16* __restrict__ Wq, const float* __restrict__ bq,
    const __bf16* __restrict__ Wk, const float* __restrict__ bk,
    const __bf16* __restrict__ Wv, const float* __restrict__ bv,
    __bf16* __restrict__ Qb, __bf16* __restrict__ Kb, __bf16* __restrict__ Vt)
{
  const int tid  = threadIdx.x;
  const int w    = tid >> 5;
  const int lane = tid & 31;
  const int ln   = lane & 15;
  const int hi   = lane >> 4;

  const int mt = blockIdx.x >> 3;                 // 32-row tile in [0, B*S/32)
  const int nt = ((blockIdx.x & 7) << 3) + w;     // 16-col tile in [0, D/16)

  const int row0 = mt * 32;                       // flat row over B*S
  const int col  = nt * 16 + ln;                  // output feature

  const __bf16* x0row = Xb + (size_t)(row0 + ln) * D_MODEL;        // M 0..15
  const __bf16* x1row = Xb + (size_t)(row0 + 16 + ln) * D_MODEL;   // M 16..31
  const __bf16* wqr   = Wq + (size_t)col * D_MODEL;                // B col = n
  const __bf16* wkr   = Wk + (size_t)col * D_MODEL;
  const __bf16* wvr   = Wv + (size_t)col * D_MODEL;

  v8f cq0 = {}, ck0 = {}, cv0 = {};
  v8f cq1 = {}, ck1 = {}, cv1 = {};

  for (int kb = 0; kb < D_MODEL / 32; ++kb) {
    const int d0 = kb * 32;
    const v16bf a0 = a_frag_bf16(x0row, d0, hi);
    const v16bf a1 = a_frag_bf16(x1row, d0, hi);
    const v16bf bqf = b_frag_bf16(wqr + d0 + 16 * hi);
    const v16bf bkf = b_frag_bf16(wkr + d0 + 16 * hi);
    const v16bf bvf = b_frag_bf16(wvr + d0 + 16 * hi);
    cq0 = wmma_bf16(a0, bqf, cq0);
    cq1 = wmma_bf16(a1, bqf, cq1);
    ck0 = wmma_bf16(a0, bkf, ck0);
    ck1 = wmma_bf16(a1, bkf, ck1);
    cv0 = wmma_bf16(a0, bvf, cv0);
    cv1 = wmma_bf16(a1, bvf, cv1);
  }

  const float biasq = bq[col];
  const float biask = bk[col];
  const float biasv = bv[col];

  const int b  = row0 / SEQ;     // 32-row tiles never straddle a batch
  const int s0 = row0 % SEQ;
  __bf16* vtb = Vt + (size_t)b * D_MODEL * SEQ + (size_t)col * SEQ;

#pragma unroll
  for (int r = 0; r < 8; ++r) {
    const int m = r + 8 * hi;
    const size_t roff0 = (size_t)(row0 + m) * D_MODEL + col;
    const size_t roff1 = (size_t)(row0 + 16 + m) * D_MODEL + col;
    Qb[roff0] = (__bf16)(cq0[r] + biasq);
    Qb[roff1] = (__bf16)(cq1[r] + biasq);
    Kb[roff0] = (__bf16)(ck0[r] + biask);
    Kb[roff1] = (__bf16)(ck1[r] + biask);
    vtb[s0 + m]      = (__bf16)(cv0[r] + biasv);   // transposed V: [d][s]
    vtb[s0 + 16 + m] = (__bf16)(cv1[r] + biasv);
  }
}

// ---------------------------------------------------------------------------
// TDM: issue an async load of one 32x1024 bf16 K tile (rows k0..k0+31) into
// LDS at byte offset lds_off.  2D tensor descriptor per CDNA5 D# layout:
//   group0: count=1, lds_addr, global_addr, type=2
//   group1: data_size=2B, tensor_dim0=D, tensor_dim1=S, tile_dim0=D,
//           tile_dim1=32, tensor_dim0_stride=D
//   remaining groups: zero (2D tensor)
// This toolchain exposes the 6-arg builtin:
//   (u32x4 g0, i32x8 g1, i32x4, i32x4, i32x8, i32 cpol)
// ---------------------------------------------------------------------------
static __device__ __forceinline__ void tdm_load_ktile(
    const __bf16* gaddr, unsigned lds_off)
{
  const unsigned long long ga = (unsigned long long)(size_t)gaddr;
  u32x4 g0;
  g0[0] = 1u;                                        // count=1, user mode
  g0[1] = lds_off;                                   // LDS byte address
  g0[2] = (unsigned)(ga & 0xffffffffu);              // global_addr[31:0]
  g0[3] = (unsigned)((ga >> 32) & 0x01ffffffu)       // global_addr[56:32]
        | (2u << 30);                                // type=2 ("image")
  i32x8 g1;
  g1[0] = 0x00010000;                                // data_size=1 (2 bytes)
  g1[1] = (int)((D_MODEL & 0xffff) << 16);           // tensor_dim0[15:0]
  g1[2] = (int)((D_MODEL >> 16) & 0xffff)            // tensor_dim0[31:16]
        | (int)((SEQ & 0xffff) << 16);               // tensor_dim1[15:0]
  g1[3] = (int)((SEQ >> 16) & 0xffff)                // tensor_dim1[31:16]
        | (int)((D_MODEL & 0xffff) << 16);           // tile_dim0 = D
  g1[4] = 32;                                        // tile_dim1=32, tile_dim2=0
  g1[5] = D_MODEL;                                   // tensor_dim0_stride[31:0]
  g1[6] = 0;                                         // stride hi / dim1_stride lo
  g1[7] = 0;
  const i32x4 gz4 = {0, 0, 0, 0};
  const i32x8 gz8 = {0, 0, 0, 0, 0, 0, 0, 0};
  __builtin_amdgcn_tensor_load_to_lds(g0, g1, gz4, gz4, gz8, 0);
}

// ---------------------------------------------------------------------------
// Kernel 2: flash attention. One block (8 waves) per (batch, 16-query tile).
// Dynamic LDS: 2 x 64 KB double-buffered K tiles (TDM destination).
// ---------------------------------------------------------------------------
__global__ __launch_bounds__(256) void attn_kernel(
    const __bf16* __restrict__ Qb, const __bf16* __restrict__ Kb,
    const __bf16* __restrict__ Vt, const float* __restrict__ pad,
    float* __restrict__ out)
{
  __shared__ float  s_scores[16 * 32];
  __shared__ __bf16 s_p[16 * 32];
  __shared__ float  s_alpha[16];
  __shared__ float  s_rsum[16];
  extern __shared__ __bf16 s_ktile[];   // [2][32][D_MODEL]

  const int tid  = threadIdx.x;
  const int w    = tid >> 5;
  const int lane = tid & 31;
  const int ln   = lane & 15;
  const int hi   = lane >> 4;

  const int qi = blockIdx.x & 127;   // S/16 = 128 query tiles
  const int b  = blockIdx.x >> 7;
  const int q0 = qi * 16;

  const __bf16* Kbb  = Kb  + (size_t)b * SEQ * D_MODEL;
  const __bf16* Vtb  = Vt  + (size_t)b * D_MODEL * SEQ;
  const float*  padb = pad + (size_t)b * SEQ;

  // LDS byte offset of the dynamic K-tile buffers (after static LDS).
  const unsigned ktile_lds0 = __builtin_amdgcn_groupstaticsize();

  // Register-resident Q fragments for this wave's 128-wide d-slice.
  const __bf16* qrow = Qb + ((size_t)b * SEQ + q0 + ln) * D_MODEL;
  v16bf qa[4];
#pragma unroll
  for (int kk = 0; kk < 4; ++kk) qa[kk] = a_frag_bf16(qrow, w * 128 + kk * 32, hi);

  v8f acc[8] = {};                  // 16 x 128 f32 output slice per wave
  float m_st = -1e30f, l_st = 0.f;  // online softmax state (lanes tid<16)

  const int nkb = (q0 + 15) / 32 + 1;   // causal: stop at the diagonal

  // Prologue: TDM-load key block 0 into buffer 0 (wave 0 issues; EXEC is
  // ignored by tensor ops, descriptors are wave-uniform SGPRs).
  if (w == 0) tdm_load_ktile(Kbb, ktile_lds0);

  for (int kbi = 0; kbi < nkb; ++kbi) {
    const int k0  = kbi * 32;
    const int buf = kbi & 1;
    const __bf16* kt = s_ktile + (size_t)buf * 32 * D_MODEL;

    // Prefetch next block's V columns (global_prefetch_b8).
    __builtin_prefetch(Vtb + (size_t)(w * 128 + ln) * SEQ + k0 + 32, 0, 1);

    // --- zero the score reduction buffer while the TDM pipeline runs
    for (int i = tid; i < 16 * 32; i += 256) s_scores[i] = 0.f;

    // Wave 0: issue next K tile, then wait until the current one is resident.
    // TDM ops from one wave complete in order, so <=1 outstanding means the
    // tile for this iteration has landed in LDS.
    if (w == 0) {
      if (kbi + 1 < nkb) {
        tdm_load_ktile(Kbb + (size_t)(k0 + 32) * D_MODEL,
                       ktile_lds0 + (unsigned)(buf ^ 1) * KTILE_BYTES);
        __builtin_amdgcn_s_wait_tensorcnt(1);
      } else {
        __builtin_amdgcn_s_wait_tensorcnt(0);
      }
    }
    __syncthreads();   // publish K tile + zeroed scores to all waves

    // --- partial scores over this wave's d-range [w*128, w*128+128)
    v8f cs0 = {}, cs1 = {};
#pragma unroll
    for (int kk = 0; kk < 4; ++kk) {
      const int d0 = w * 128 + kk * 32;
      // B-fragments from the LDS-resident K tile (ds_load_b128 pairs).
      v16bf b0, b1;
      {
        bf16x8 c0 = *(const bf16x8*)&kt[(size_t)ln * D_MODEL + d0 + 16 * hi];
        bf16x8 c1 = *(const bf16x8*)&kt[(size_t)ln * D_MODEL + d0 + 16 * hi + 8];
        bf16x8 c2 = *(const bf16x8*)&kt[(size_t)(16 + ln) * D_MODEL + d0 + 16 * hi];
        bf16x8 c3 = *(const bf16x8*)&kt[(size_t)(16 + ln) * D_MODEL + d0 + 16 * hi + 8];
#pragma unroll
        for (int i = 0; i < 8; ++i) {
          b0[i] = c0[i]; b0[i + 8] = c1[i];
          b1[i] = c2[i]; b1[i + 8] = c3[i];
        }
      }
      cs0 = wmma_bf16(qa[kk], b0, cs0);
      cs1 = wmma_bf16(qa[kk], b1, cs1);
    }
#pragma unroll
    for (int r = 0; r < 8; ++r) {
      atomicAdd(&s_scores[(r + 8 * hi) * 32 + ln], cs0[r]);
      atomicAdd(&s_scores[(r + 8 * hi) * 32 + 16 + ln], cs1[r]);
    }
    __syncthreads();

    // --- online softmax update (16 rows handled by threads 0..15)
    if (tid < 16) {
      const int row = tid;
      const int qg  = q0 + row;
      float rmax = m_st;
#pragma unroll 8
      for (int c = 0; c < 32; ++c) {
        const int kg = k0 + c;
        const float v = s_scores[row * 32 + c] * ATT_SCALE + padb[kg] +
                        ((kg > qg) ? NEG_BIG : 0.f);
        rmax = fmaxf(rmax, v);
      }
      const float al = __expf(m_st - rmax);
      float ps = 0.f;
#pragma unroll 8
      for (int c = 0; c < 32; ++c) {
        const int kg = k0 + c;
        const float v = s_scores[row * 32 + c] * ATT_SCALE + padb[kg] +
                        ((kg > qg) ? NEG_BIG : 0.f);
        const float p = __expf(v - rmax);
        ps += p;
        s_p[row * 32 + c] = (__bf16)p;
      }
      l_st = l_st * al + ps;
      m_st = rmax;
      s_alpha[row] = al;
    }
    __syncthreads();

    // --- rescale accumulators by alpha(row)
    float alr[8];
#pragma unroll
    for (int r = 0; r < 8; ++r) alr[r] = s_alpha[8 * hi + r];
#pragma unroll
    for (int t = 0; t < 8; ++t)
#pragma unroll
      for (int r = 0; r < 8; ++r) acc[t][r] *= alr[r];

    // --- P(16x32) * V(32x128 slice): A fragment from LDS, B from Vt (contiguous)
    v16bf pa;
    {
      bf16x8 p0 = *(const bf16x8*)&s_p[ln * 32 + 16 * hi];
      bf16x8 p1 = *(const bf16x8*)&s_p[ln * 32 + 16 * hi + 8];
#pragma unroll
      for (int i = 0; i < 8; ++i) { pa[i] = p0[i]; pa[i + 8] = p1[i]; }
    }
#pragma unroll
    for (int t = 0; t < 8; ++t) {
      const int col = w * 128 + t * 16 + ln;
      const v16bf vb = b_frag_bf16(Vtb + (size_t)col * SEQ + k0 + 16 * hi);
      acc[t] = wmma_bf16(pa, vb, acc[t]);
    }
    __syncthreads();   // readers done before buf^1 is overwritten next iter
  }

  // --- final 1/rowsum normalization and fp32 store
  if (tid < 16) s_rsum[tid] = l_st;
  __syncthreads();
  float rinv[8];
#pragma unroll
  for (int r = 0; r < 8; ++r) rinv[r] = 1.f / s_rsum[8 * hi + r];

#pragma unroll
  for (int t = 0; t < 8; ++t) {
    const int col = w * 128 + t * 16 + ln;
#pragma unroll
    for (int r = 0; r < 8; ++r) {
      const int m = r + 8 * hi;
      out[(size_t)b * SEQ * D_MODEL + (size_t)(q0 + m) * D_MODEL + col] =
          acc[t][r] * rinv[r];
    }
  }
}

// ---------------------------------------------------------------------------
extern "C" void kernel_launch(void* const* d_in, const int* in_sizes, int n_in,
                              void* d_out, int out_size, void* d_ws, size_t ws_size,
                              hipStream_t stream) {
  (void)in_sizes; (void)n_in; (void)out_size; (void)ws_size;
  // setup_inputs order: src, src_padding_mask, src_subsq_mask, Wk, bk, Wv, bv, Wq, bq
  const float* src = (const float*)d_in[0];
  const float* pad = (const float*)d_in[1];   // [B,S,1] additive mask, flat B*S
  // d_in[2] (causal mask) is reproduced analytically in-kernel
  const float* Wk = (const float*)d_in[3];
  const float* bk = (const float*)d_in[4];
  const float* Wv = (const float*)d_in[5];
  const float* bv = (const float*)d_in[6];
  const float* Wq = (const float*)d_in[7];
  const float* bq = (const float*)d_in[8];
  float* out = (float*)d_out;

  const long N  = (long)BATCH * SEQ * D_MODEL;   // 16M elements
  const long NW = (long)D_MODEL * D_MODEL;       // 1M elements per weight

  __bf16* xb  = (__bf16*)d_ws;     // bf16 src              (32 MB)
  __bf16* wqb = xb  + N;           // bf16 weights          (3 x 2 MB)
  __bf16* wkb = wqb + NW;
  __bf16* wvb = wkb + NW;
  __bf16* qb  = wvb + NW;          // bf16 Q,K              (2 x 32 MB)
  __bf16* kb  = qb  + N;
  __bf16* vt  = kb  + N;           // bf16 V transposed [b][d][s] (32 MB)

  // One-time fp32 -> bf16 conversions (memory bound).
  cvt_bf16_kernel<<<dim3((unsigned)(N  / (256 * 8))), dim3(256), 0, stream>>>(src, xb,  N);
  cvt_bf16_kernel<<<dim3((unsigned)(NW / (256 * 8))), dim3(256), 0, stream>>>(Wq,  wqb, NW);
  cvt_bf16_kernel<<<dim3((unsigned)(NW / (256 * 8))), dim3(256), 0, stream>>>(Wk,  wkb, NW);
  cvt_bf16_kernel<<<dim3((unsigned)(NW / (256 * 8))), dim3(256), 0, stream>>>(Wv,  wvb, NW);

  // M tiles = B*S/32 = 512, N tile groups = (D/16)/8 = 8  -> 4096 blocks
  qkv_proj_kernel<<<dim3((BATCH * SEQ / 32) * 8), dim3(256), 0, stream>>>(
      xb, wqb, bq, wkb, bk, wvb, bv, qb, kb, vt);

  // one block per (batch, 16-query tile); 128 KB dynamic LDS for the
  // double-buffered TDM K tiles.
  attn_kernel<<<dim3(BATCH * (SEQ / 16)), dim3(256), 2 * KTILE_BYTES, stream>>>(
      qb, kb, vt, pad, out);
}
